// SuperGlue_89739046683476
// MI455X (gfx1250) — compile-verified, matrix operand order checked by reading, not run on hardware
//
#include <hip/hip_runtime.h>
#include <hip/hip_bf16.h>
#include <math.h>

typedef __bf16 bf16;
typedef bf16  v16bf __attribute__((ext_vector_type(16)));
typedef bf16  v8bf  __attribute__((ext_vector_type(8)));
typedef float v8f   __attribute__((ext_vector_type(8)));

#define TOK    1024
#define DCH    256
#define NS     4          // streams: batch(2) x image(2), z = batch*2 + image
#define NLAYER 18
#define MP     1025       // sinkhorn matrix dim (N+1)
#define NBAT   18         // output batches: 9 cross layers x 2
#define LDT    48         // padded LDS tile row stride (bf16 elems), 96B (16B-aligned rows)

// ---------------------------------------------------------------- fragments
__device__ __forceinline__ v16bf cat8(v8bf lo, v8bf hi){
  return __builtin_shufflevector(lo, hi, 0,1,2,3,4,5,6,7,8,9,10,11,12,13,14,15);
}
// Load a 16-lane-row WMMA bf16 fragment slice. Caller passes
// p = row_base + hf*8 (elements), hf = lane>>4. Elements 0..7 <- K=hf*8+0..7,
// elements 8..15 <- K=16+hf*8+0..7 (per CDNA5 16-bit A/B 16x32 layout).
__device__ __forceinline__ v16bf ldfrag(const bf16* p){
  v8bf lo = *(const v8bf*)p;
  v8bf hi = *(const v8bf*)(p + 16);
  return cat8(lo, hi);
}
__device__ __forceinline__ float red16_max(float x){
#pragma unroll
  for (int m = 1; m < 16; m <<= 1) x = fmaxf(x, __shfl_xor(x, m, 32));
  return x;
}
__device__ __forceinline__ float red16_sum(float x){
#pragma unroll
  for (int m = 1; m < 16; m <<= 1) x += __shfl_xor(x, m, 32);
  return x;
}

// ---------------------------------------------------------------- generic WMMA GEMM
// C[i,j] = sum_k A[zA][i*lda+k] * B[z][j*ldb+k]   (both operands K-contiguous)
// EPI 0: bf16 out (+bias)                EPI 1: bf16 out transposed (+bias)
// EPI 2: bf16 out, y=acc*scale+shift, relu
// EPI 3: residual: resF += acc+bias; resB = bf16(resF)   EPI 4: f32 out * out_scale
template<int EPI>
__global__ __launch_bounds__(128) void gemm_wmma(
    const bf16* __restrict__ A, const bf16* __restrict__ B, void* __restrict__ Cout,
    const float* __restrict__ bias, const float* __restrict__ scale,
    const float* __restrict__ shift,
    float* __restrict__ resF, bf16* __restrict__ resB,
    int M, int N, int K, int lda, int ldb, int ldc,
    long long sA, long long sB, long long sC, int a_xor, float out_scale)
{
  __shared__ __align__(16) bf16 At[64*LDT];
  __shared__ __align__(16) bf16 Bt[64*LDT];
  const int z  = blockIdx.z;
  const bf16* Ab = A + (long long)(z ^ a_xor) * sA;
  const bf16* Bb = B + (long long)z * sB;
  const int m0 = blockIdx.x*64, n0 = blockIdx.y*64;
  const int tid = threadIdx.x, wv = tid>>5, ln = tid&31;
  const int hf = ln>>4, lm = ln&15;
  const int wm = (wv>>1)*32, wn = (wv&1)*32;
  v8f acc[2][2] = {};
  for (int kb = 0; kb < K; kb += 32){
    for (int c = tid; c < 256; c += 128){        // 2 x 16B chunks per thread per tile
      int r = c >> 2, co = (c & 3) * 8;
      *(uint4*)&At[r*LDT + co] = *(const uint4*)&Ab[(long long)(m0 + r)*lda + kb + co];
      *(uint4*)&Bt[r*LDT + co] = *(const uint4*)&Bb[(long long)(n0 + r)*ldb + kb + co];
    }
    if (kb + 32 < K){
      __builtin_prefetch(&Ab[(long long)(m0 + (tid>>1))*lda + kb + 32], 0, 0);
      __builtin_prefetch(&Bb[(long long)(n0 + (tid>>1))*ldb + kb + 32], 0, 0);
    }
    __syncthreads();
    v16bf af[2], bfr[2];
#pragma unroll
    for (int i = 0; i < 2; i++){
      af[i]  = ldfrag(&At[(wm + i*16 + lm)*LDT + hf*8]);
      bfr[i] = ldfrag(&Bt[(wn + i*16 + lm)*LDT + hf*8]);
    }
#pragma unroll
    for (int mi = 0; mi < 2; mi++)
#pragma unroll
      for (int ni = 0; ni < 2; ni++)
        acc[mi][ni] = __builtin_amdgcn_wmma_f32_16x16x32_bf16(
            false, af[mi], false, bfr[ni], (short)0, acc[mi][ni], false, false);
    __syncthreads();
  }
#pragma unroll
  for (int mi = 0; mi < 2; mi++){
#pragma unroll
    for (int ni = 0; ni < 2; ni++){
#pragma unroll
      for (int r = 0; r < 8; r++){
        int gm = m0 + wm + mi*16 + 8*hf + r;     // D-layout: row = vgpr + 8*(lane/16)
        int gn = n0 + wn + ni*16 + lm;           //           col = lane%16
        float v = acc[mi][ni][r];
        if constexpr (EPI == 0){
          v += bias ? bias[gn] : 0.f;
          ((bf16*)Cout)[(long long)z*sC + (long long)gm*ldc + gn] = (bf16)v;
        } else if constexpr (EPI == 1){
          v += bias ? bias[gn] : 0.f;
          ((bf16*)Cout)[(long long)z*sC + (long long)gn*ldc + gm] = (bf16)v;
        } else if constexpr (EPI == 2){
          v = fmaxf(v*scale[gn] + shift[gn], 0.f);
          ((bf16*)Cout)[(long long)z*sC + (long long)gm*ldc + gn] = (bf16)v;
        } else if constexpr (EPI == 3){
          v += bias ? bias[gn] : 0.f;
          long long idx = (long long)z*sC + (long long)gm*ldc + gn;
          float nv = resF[idx] + v;
          resF[idx] = nv;
          resB[idx] = (bf16)nv;
        } else {
          ((float*)Cout)[(long long)z*sC + (long long)gm*ldc + gn] = v*out_scale;
        }
      }
    }
  }
}

// ---------------------------------------------------------------- flash attention
// q,k: [NS][TOK][256] bf16 head-major (c = h*64+d); vt: [NS][256][TOK] bf16
// out: [NS][TOK][256] bf16 head-major. One wave = 16 query rows, 32-key steps.
__global__ __launch_bounds__(128) void flash_attn(
    const bf16* __restrict__ q, const bf16* __restrict__ k,
    const bf16* __restrict__ vt, bf16* __restrict__ out, float sm_scale)
{
  __shared__ __align__(16) bf16 pst[4*16*LDT];   // per-wave P staging (D->A relayout)
  const int z = blockIdx.z, h = blockIdx.y;
  const int wv = threadIdx.x>>5, ln = threadIdx.x&31, hf = ln>>4, lm = ln&15;
  const int qbase = blockIdx.x*64 + wv*16;
  const bf16* qz = q  + (long long)z*TOK*DCH;
  const bf16* kz = k  + (long long)z*TOK*DCH;
  const bf16* vz = vt + (long long)z*DCH*TOK;
  bf16* pw = &pst[wv*16*LDT];

  v16bf qf[2];
#pragma unroll
  for (int kc = 0; kc < 2; kc++)
    qf[kc] = ldfrag(qz + (long long)(qbase + lm)*DCH + h*64 + kc*32 + hf*8);

  float mrow[8], lrow[8];
  v8f oacc[4] = {};
#pragma unroll
  for (int r = 0; r < 8; r++){ mrow[r] = -INFINITY; lrow[r] = 0.f; }

  for (int jb = 0; jb < TOK; jb += 32){
    v8f s[2] = {};
#pragma unroll
    for (int nt = 0; nt < 2; nt++)
#pragma unroll
      for (int kc = 0; kc < 2; kc++){
        v16bf kf = ldfrag(kz + (long long)(jb + nt*16 + lm)*DCH + h*64 + kc*32 + hf*8);
        s[nt] = __builtin_amdgcn_wmma_f32_16x16x32_bf16(
            false, qf[kc], false, kf, (short)0, s[nt], false, false);
      }
    float mnew[8], alpha[8];
#pragma unroll
    for (int r = 0; r < 8; r++){
      float a = s[0][r]*sm_scale, b = s[1][r]*sm_scale;
      s[0][r] = a; s[1][r] = b;
      float tm = red16_max(fmaxf(a, b));
      mnew[r]  = fmaxf(mrow[r], tm);
      alpha[r] = __expf(mrow[r] - mnew[r]);
      mrow[r]  = mnew[r];
    }
#pragma unroll
    for (int r = 0; r < 8; r++){
      float p0 = __expf(s[0][r] - mnew[r]);
      float p1 = __expf(s[1][r] - mnew[r]);
      s[0][r] = p0; s[1][r] = p1;
      lrow[r] = lrow[r]*alpha[r] + red16_sum(p0 + p1);
    }
#pragma unroll
    for (int dt = 0; dt < 4; dt++)
#pragma unroll
      for (int r = 0; r < 8; r++) oacc[dt][r] *= alpha[r];
    // D-layout -> LDS [16 q][32 keys] (same-wave LDS is in-order)
#pragma unroll
    for (int nt = 0; nt < 2; nt++)
#pragma unroll
      for (int r = 0; r < 8; r++)
        pw[(8*hf + r)*LDT + nt*16 + lm] = (bf16)s[nt][r];
    v16bf pf = ldfrag(&pw[lm*LDT + hf*8]);       // reload as A-fragment
#pragma unroll
    for (int dt = 0; dt < 4; dt++){
      v16bf vf = ldfrag(vz + (long long)(h*64 + dt*16 + lm)*TOK + jb + hf*8);
      oacc[dt] = __builtin_amdgcn_wmma_f32_16x16x32_bf16(
          false, pf, false, vf, (short)0, oacc[dt], false, false);
    }
  }
  bf16* ob = out + (long long)z*TOK*DCH;
#pragma unroll
  for (int r = 0; r < 8; r++){
    float inv = 1.f / lrow[r];
    int row = qbase + 8*hf + r;
#pragma unroll
    for (int dt = 0; dt < 4; dt++)
      ob[(long long)row*DCH + h*64 + dt*16 + lm] = (bf16)(oacc[dt][r]*inv);
  }
}

// ---------------------------------------------------------------- small kernels
__global__ void k_f32_to_bf16(const float* __restrict__ s, bf16* __restrict__ d, int n){
  int i = blockIdx.x*blockDim.x + threadIdx.x;
  if (i < n) d[i] = (bf16)s[i];
}
// reference reshape (b, HDIM, HEADS, N): channel c = d*4 + h -> head-major h*64+d
__global__ void k_conv_qkv_w(const float* __restrict__ s, bf16* __restrict__ d){
  int i = blockIdx.x*blockDim.x + threadIdx.x;   // 65536
  if (i >= 65536) return;
  int op = i >> 8, c = i & 255;
  int h = op >> 6, dd = op & 63;
  d[i] = (bf16)s[((dd<<2) + h)*256 + c];
}
__global__ void k_perm_bias(const float* __restrict__ s, float* __restrict__ d){
  int op = threadIdx.x;                          // 256
  int h = op >> 6, dd = op & 63;
  d[op] = s[(dd<<2) + h];
}
__global__ void k_conv_merge_w(const float* __restrict__ s, bf16* __restrict__ d){
  int i = blockIdx.x*blockDim.x + threadIdx.x;   // 65536
  if (i >= 65536) return;
  int o = i >> 8, cp = i & 255;
  int h = cp >> 6, dd = cp & 63;
  d[i] = (bf16)s[o*256 + (dd<<2) + h];
}
__global__ void k_fuse_bn(const float* __restrict__ b, const float* __restrict__ g,
                          const float* __restrict__ be,
                          float* __restrict__ sc, float* __restrict__ sh, int n){
  int i = blockIdx.x*blockDim.x + threadIdx.x;
  if (i < n){
    float s = g[i] * 0.9999950000374997f;        // 1/sqrt(1+1e-5)
    sc[i] = s;
    sh[i] = b[i]*s + be[i];
  }
}
__global__ void k_build_kin(const float* __restrict__ kp0, const float* __restrict__ kp1,
                            const float* __restrict__ sc0, const float* __restrict__ sc1,
                            float* __restrict__ kin){
  int t = blockIdx.x*blockDim.x + threadIdx.x;
  if (t >= NS*TOK) return;
  int z = t / TOK, n = t % TOK;
  int img = z & 1, bat = z >> 1;
  const float* kp = img ? kp1 : kp0;
  const float* sc = img ? sc1 : sc0;
  kin[t*3+0] = kp[((long long)bat*TOK + n)*2 + 0];
  kin[t*3+1] = kp[((long long)bat*TOK + n)*2 + 1];
  kin[t*3+2] = sc[(long long)bat*TOK + n];
}
__global__ void k_mlp_stage(const float* __restrict__ src, const float* __restrict__ w,
                            const float* __restrict__ scale, const float* __restrict__ shift,
                            float* __restrict__ dst, int Cin, int Cout){
  int idx = blockIdx.x*blockDim.x + threadIdx.x;
  if (idx >= NS*TOK*Cout) return;
  int t = idx / Cout, o = idx % Cout;
  const float* s  = src + (long long)t*Cin;
  const float* wr = w   + (long long)o*Cin;
  float acc = 0.f;
  for (int i = 0; i < Cin; i++) acc += s[i]*wr[i];
  dst[idx] = fmaxf(acc*scale[o] + shift[o], 0.f);
}
__global__ void k_kenc_final(const float* __restrict__ src, const float* __restrict__ w,
                             const float* __restrict__ b,
                             const float* __restrict__ d0, const float* __restrict__ d1,
                             float* __restrict__ xf, bf16* __restrict__ xb){
  int idx = blockIdx.x*blockDim.x + threadIdx.x;
  if (idx >= NS*TOK*DCH) return;
  int t = idx / DCH, o = idx % DCH;
  int z = t / TOK, n = t % TOK;
  int img = z & 1, bat = z >> 1;
  const float* s  = src + (long long)t*128;
  const float* wr = w   + (long long)o*128;
  float acc = b[o];
  for (int i = 0; i < 128; i++) acc += s[i]*wr[i];
  const float* dd = img ? d1 : d0;
  float v = dd[((long long)bat*TOK + n)*DCH + o] + acc;
  xf[idx] = v;
  xb[idx] = (bf16)v;
}
__global__ void k_copy_cat(const bf16* __restrict__ xb, bf16* __restrict__ cat){
  int idx = blockIdx.x*blockDim.x + threadIdx.x;
  if (idx >= NS*TOK*DCH) return;
  int t = idx / DCH, c = idx % DCH;
  cat[(long long)t*512 + c] = xb[idx];
}
__global__ void k_bin_fill(float* __restrict__ C, const float* __restrict__ bin){
  int idx = blockIdx.x*blockDim.x + threadIdx.x;
  if (idx >= NBAT*MP) return;
  float a = *bin;
  int bi = idx / MP, t = idx % MP;
  float* Cb = C + (size_t)bi*MP*MP;
  Cb[(size_t)1024*MP + t] = a;
  Cb[(size_t)t*MP + 1024] = a;
}
__global__ void k_zero(float* p, int n){
  int i = blockIdx.x*blockDim.x + threadIdx.x;
  if (i < n) p[i] = 0.f;
}
__global__ __launch_bounds__(256) void k_sink_row(const float* __restrict__ C,
                                                  const float* __restrict__ v,
                                                  float* __restrict__ u){
  int bi = blockIdx.y, i = blockIdx.x;
  const float* row = C + ((size_t)bi*MP + i)*MP;
  const float* vb  = v + bi*MP;
  __shared__ float red[256];
  int t = threadIdx.x;
  float m = -INFINITY;
  for (int j = t; j < MP; j += 256) m = fmaxf(m, row[j] + vb[j]);
  red[t] = m; __syncthreads();
  for (int s = 128; s > 0; s >>= 1){ if (t < s) red[t] = fmaxf(red[t], red[t+s]); __syncthreads(); }
  m = red[0]; __syncthreads();
  float sum = 0.f;
  for (int j = t; j < MP; j += 256) sum += __expf(row[j] + vb[j] - m);
  red[t] = sum; __syncthreads();
  for (int s = 128; s > 0; s >>= 1){ if (t < s) red[t] += red[t+s]; __syncthreads(); }
  if (t == 0){
    float lse  = m + __logf(red[0]);
    float norm = -__logf(2048.f);
    float lmu  = (i < 1024) ? norm : (__logf(1024.f) + norm);
    u[bi*MP + i] = lmu - lse;
  }
}
__global__ __launch_bounds__(256) void k_sink_col(const float* __restrict__ C,
                                                  const float* __restrict__ u,
                                                  float* __restrict__ v){
  int bi = blockIdx.y, j = blockIdx.x;
  const float* Cb = C + (size_t)bi*MP*MP;
  const float* ub = u + bi*MP;
  __shared__ float red[256];
  int t = threadIdx.x;
  float m = -INFINITY;
  for (int i = t; i < MP; i += 256) m = fmaxf(m, Cb[(size_t)i*MP + j] + ub[i]);
  red[t] = m; __syncthreads();
  for (int s = 128; s > 0; s >>= 1){ if (t < s) red[t] = fmaxf(red[t], red[t+s]); __syncthreads(); }
  m = red[0]; __syncthreads();
  float sum = 0.f;
  for (int i = t; i < MP; i += 256) sum += __expf(Cb[(size_t)i*MP + j] + ub[i] - m);
  red[t] = sum; __syncthreads();
  for (int s = 128; s > 0; s >>= 1){ if (t < s) red[t] += red[t+s]; __syncthreads(); }
  if (t == 0){
    float lse  = m + __logf(red[0]);
    float norm = -__logf(2048.f);
    float lnu  = (j < 1024) ? norm : (__logf(1024.f) + norm);
    v[bi*MP + j] = lnu - lse;
  }
}
__global__ void k_sink_final(float* __restrict__ C, const float* __restrict__ u,
                             const float* __restrict__ v){
  size_t idx = (size_t)blockIdx.x*blockDim.x + threadIdx.x;
  size_t total = (size_t)NBAT*MP*MP;
  if (idx >= total) return;
  size_t bi = idx / ((size_t)MP*MP);
  size_t rem = idx % ((size_t)MP*MP);
  int i = (int)(rem / MP), j = (int)(rem % MP);
  float norm = -__logf(2048.f);
  C[idx] += u[bi*MP + i] + v[bi*MP + j] - norm;
}

// ---------------------------------------------------------------- host
#define GEMM(EPI, A,B,C,BIAS,SC,SH,RF,RB, M,N,K, LDA,LDB,LDC, SA,SB_,SC_, AX,OS, BAT) \
  gemm_wmma<EPI><<<dim3((M)/64,(N)/64,(BAT)), dim3(128), 0, stream>>>(               \
      A,B,C,BIAS,SC,SH,RF,RB, M,N,K, LDA,LDB,LDC, SA,SB_,SC_, AX,OS)

extern "C" void kernel_launch(void* const* d_in, const int* in_sizes, int n_in,
                              void* d_out, int out_size, void* d_ws, size_t ws_size,
                              hipStream_t stream)
{
  (void)in_sizes; (void)n_in; (void)out_size; (void)ws_size;
  // input map (jax pytree leaves, dict keys sorted):
  // 0..5 data, 6 bin_score, 7 final.b, 8 final.w,
  // per gnn layer (14): merge.b, merge.w, q.b, q.w, k.b, k.w, v.b, v.w,
  //                     bn.be, bn.g, c0.b, c0.w, c1.b, c1.w
  // kenc (14): bn0.be, bn0.g, bn1.be, bn1.g, bn2.be, bn2.g,
  //            c0.b, c0.w, c1.b, c1.w, c2.b, c2.w, c3.b, c3.w
  const float* desc0 = (const float*)d_in[0];
  const float* desc1 = (const float*)d_in[1];
  const float* kp0   = (const float*)d_in[2];
  const float* kp1   = (const float*)d_in[3];
  const float* sc0   = (const float*)d_in[4];
  const float* sc1   = (const float*)d_in[5];
  const float* bin   = (const float*)d_in[6];
  const float* fin_b = (const float*)d_in[7];
  const float* fin_w = (const float*)d_in[8];
  const int GNNB = 9, GST = 14, KEB = 9 + NLAYER*GST;
  auto L  = [&](int l, int j){ return (const float*)d_in[GNNB + l*GST + j]; };
  auto KE = [&](int j){ return (const float*)d_in[KEB + j]; };

  size_t off = 0;
  auto alloc = [&](size_t bytes) -> void* {
    off = (off + 255) & ~(size_t)255;
    void* p = (char*)d_ws + off;
    off += bytes;
    return p;
  };
  const long long sAct = (long long)TOK*DCH;
  const size_t SBE = (size_t)NS*TOK*DCH;
  float* xf  = (float*)alloc(SBE*4);            // fp32 residual stream [NS][TOK][256]
  bf16*  xb  = (bf16*) alloc(SBE*2);            // bf16 mirror
  bf16*  qb  = (bf16*) alloc(SBE*2);
  bf16*  kb  = (bf16*) alloc(SBE*2);
  bf16*  vtb = (bf16*) alloc(SBE*2);            // V transposed [NS][256][TOK]
  bf16*  ab  = (bf16*) alloc(SBE*2);            // attention output
  bf16*  cat = (bf16*) alloc(SBE*2*2);          // [NS][TOK][512] = [x | message]
  bf16*  hb  = (bf16*) alloc(SBE*2*2);          // mlp hidden [NS][TOK][512]
  bf16*  md  = (bf16*) alloc(SBE*2);            // final-proj descriptors
  float* kin = (float*)alloc((size_t)NS*TOK*3*4);
  float* e1  = (float*)alloc((size_t)NS*TOK*32*4);
  float* e2  = (float*)alloc((size_t)NS*TOK*64*4);
  float* e3  = (float*)alloc((size_t)NS*TOK*128*4);
  float* ksc = (float*)alloc(224*4);
  float* ksh = (float*)alloc(224*4);
  bf16*  wqb = (bf16*) alloc(65536*2);
  bf16*  wkb = (bf16*) alloc(65536*2);
  bf16*  wvb = (bf16*) alloc(65536*2);
  bf16*  wmb = (bf16*) alloc(65536*2);
  bf16*  w1b = (bf16*) alloc((size_t)512*512*2);
  bf16*  w2b = (bf16*) alloc((size_t)256*512*2);
  float* bqp = (float*)alloc(256*4);
  float* bkp = (float*)alloc(256*4);
  float* bvp = (float*)alloc(256*4);
  float* msc = (float*)alloc(512*4);
  float* msh = (float*)alloc(512*4);
  bf16*  wfb = (bf16*) alloc(65536*2);
  float* uu  = (float*)alloc((size_t)NBAT*MP*4);
  float* vv  = (float*)alloc((size_t)NBAT*MP*4);

  // ---- keypoint encoder (fp32, tiny) + residual init
  k_fuse_bn<<<1, 64, 0, stream>>>(KE(6),  KE(1), KE(0), ksc+0,  ksh+0,  32);
  k_fuse_bn<<<1, 64, 0, stream>>>(KE(8),  KE(3), KE(2), ksc+32, ksh+32, 64);
  k_fuse_bn<<<1, 128,0, stream>>>(KE(10), KE(5), KE(4), ksc+96, ksh+96, 128);
  k_build_kin<<<(NS*TOK+255)/256, 256, 0, stream>>>(kp0, kp1, sc0, sc1, kin);
  k_mlp_stage<<<(NS*TOK*32 +255)/256, 256, 0, stream>>>(kin, KE(7),  ksc+0,  ksh+0,  e1, 3,  32);
  k_mlp_stage<<<(NS*TOK*64 +255)/256, 256, 0, stream>>>(e1,  KE(9),  ksc+32, ksh+32, e2, 32, 64);
  k_mlp_stage<<<(NS*TOK*128+255)/256, 256, 0, stream>>>(e2,  KE(11), ksc+96, ksh+96, e3, 64, 128);
  k_kenc_final<<<(NS*TOK*DCH+255)/256, 256, 0, stream>>>(e3, KE(13), KE(12), desc0, desc1, xf, xb);

  k_f32_to_bf16<<<(65536+255)/256, 256, 0, stream>>>(fin_w, wfb, 65536);

  // ---- GNN layers
  for (int l = 0; l < NLAYER; l++){
    int cross = l & 1;
    k_conv_qkv_w<<<256, 256, 0, stream>>>(L(l,3), wqb);
    k_conv_qkv_w<<<256, 256, 0, stream>>>(L(l,5), wkb);
    k_conv_qkv_w<<<256, 256, 0, stream>>>(L(l,7), wvb);
    k_perm_bias<<<1, 256, 0, stream>>>(L(l,2), bqp);
    k_perm_bias<<<1, 256, 0, stream>>>(L(l,4), bkp);
    k_perm_bias<<<1, 256, 0, stream>>>(L(l,6), bvp);
    k_conv_merge_w<<<256, 256, 0, stream>>>(L(l,1), wmb);
    k_fuse_bn<<<2, 256, 0, stream>>>(L(l,10), L(l,9), L(l,8), msc, msh, 512);
    k_f32_to_bf16<<<((512*512)+255)/256, 256, 0, stream>>>(L(l,11), w1b, 512*512);
    k_f32_to_bf16<<<((256*512)+255)/256, 256, 0, stream>>>(L(l,13), w2b, 256*512);

    GEMM(0, xb, wqb, qb,  bqp, nullptr, nullptr, nullptr, nullptr,
         TOK, DCH, DCH, DCH, DCH, DCH, sAct, 0, sAct, 0,     1.f, NS);
    GEMM(0, xb, wkb, kb,  bkp, nullptr, nullptr, nullptr, nullptr,
         TOK, DCH, DCH, DCH, DCH, DCH, sAct, 0, sAct, cross, 1.f, NS);
    GEMM(1, xb, wvb, vtb, bvp, nullptr, nullptr, nullptr, nullptr,
         TOK, DCH, DCH, DCH, DCH, TOK, sAct, 0, sAct, cross, 1.f, NS);
    flash_attn<<<dim3(TOK/64, 4, NS), dim3(128), 0, stream>>>(qb, kb, vtb, ab, 0.125f);
    k_copy_cat<<<(NS*TOK*DCH+255)/256, 256, 0, stream>>>(xb, cat);
    GEMM(0, ab, wmb, cat + DCH, L(l,0), nullptr, nullptr, nullptr, nullptr,
         TOK, DCH, DCH, DCH, DCH, 512, sAct, 0, (long long)TOK*512, 0, 1.f, NS);
    GEMM(2, cat, w1b, hb, nullptr, msc, msh, nullptr, nullptr,
         TOK, 512, 512, 512, 512, 512, (long long)TOK*512, 0, (long long)TOK*512, 0, 1.f, NS);
    GEMM(3, hb, w2b, nullptr, L(l,12), nullptr, nullptr, xf, xb,
         TOK, DCH, 512, 512, 512, DCH, (long long)TOK*512, 0, sAct, 0, 1.f, NS);

    if (cross){
      int lc = l >> 1;
      GEMM(0, xb, wfb, md, fin_b, nullptr, nullptr, nullptr, nullptr,
           TOK, DCH, DCH, DCH, DCH, DCH, sAct, 0, sAct, 0, 1.f, NS);
      float* Cc = (float*)d_out + (size_t)(lc*2)*MP*MP;
      GEMM(4, md, md + sAct, Cc, nullptr, nullptr, nullptr, nullptr, nullptr,
           TOK, TOK, DCH, DCH, DCH, MP, 2*sAct, 2*sAct, (long long)MP*MP, 0, 0.0625f, 2);
    }
  }

  // ---- Sinkhorn in log space (couplings live in d_out)
  k_bin_fill<<<(NBAT*MP+255)/256, 256, 0, stream>>>((float*)d_out, bin);
  k_zero<<<(NBAT*MP+255)/256, 256, 0, stream>>>(uu, NBAT*MP);
  k_zero<<<(NBAT*MP+255)/256, 256, 0, stream>>>(vv, NBAT*MP);
  for (int it = 0; it < 20; it++){
    k_sink_row<<<dim3(MP, NBAT), 256, 0, stream>>>((const float*)d_out, vv, uu);
    k_sink_col<<<dim3(MP, NBAT), 256, 0, stream>>>((const float*)d_out, uu, vv);
  }
  size_t total = (size_t)NBAT*MP*MP;
  k_sink_final<<<(unsigned)((total + 255)/256), 256, 0, stream>>>((float*)d_out, uu, vv);
}
